// RNF_12833362280957
// MI455X (gfx1250) — matrix operand order repaired and checked by reading, not compiled, not run on hardware
//
#include <hip/hip_runtime.h>

// ---------------------------------------------------------------------------
// MI455X (gfx1250) fused conv-window LSTM classifier.
//
//   K0: convert w_ih / w_hh f32 -> bf16 blobs in workspace (TDM source)
//   K1: P[b,s,0:512] = bf16(emb[tok[b,s]]) @ bf16(w_ih)^T + (b_ih+b_hh)   (WMMA)
//   K2: 6-step LSTM over N=32448 rows; g = P_row + h @ w_hh^T (WMMA, w_hh in
//       LDS via TDM tensor_load_to_lds with hardware pad-insertion)
//   K3: max over W=507 windows + 128->2 FC
// ---------------------------------------------------------------------------

typedef __attribute__((ext_vector_type(16))) __bf16 v16bf;
typedef __attribute__((ext_vector_type(8)))  __bf16 v8bf;
typedef __attribute__((ext_vector_type(8)))  float  v8f;
typedef __attribute__((ext_vector_type(4)))  unsigned u32x4;
typedef __attribute__((ext_vector_type(8)))  int      i32x8;
typedef __attribute__((ext_vector_type(4)))  int      i32x4;

namespace {
constexpr int kB  = 64;
constexpr int kS  = 512;
constexpr int kE  = 128;
constexpr int kH  = 128;
constexpr int kG  = 512;              // 4*H
constexpr int kFW = 6;
constexpr int kW  = kS - kFW + 1;     // 507
constexpr int kN  = kB * kW;          // 32448
// padded LDS row stride: 136 elems = 68 dwords -> bank offset 4*l mod 64,
// conflict-free across the 16 lanes of a fragment load; 16B aligned.
constexpr int kP  = 136;
constexpr int kWLDS = kG * kP;        // padded weight tile, elems (69632)
}

__device__ __forceinline__ float sigm(float x) { return 1.0f / (1.0f + __expf(-x)); }

// A-fragment (16-bit, 16x32): lane l = row (l&15); k-runs {b..b+7} u {b+16..b+23}
__device__ __forceinline__ v16bf load_a_frag(const __bf16* rowbase, int lane, int k0) {
  const __bf16* p = rowbase + k0 + ((lane & 16) ? 8 : 0);
  v8bf lo = *(const v8bf*)(p);
  v8bf hi = *(const v8bf*)(p + 16);
  return __builtin_shufflevector(lo, hi, 0,1,2,3,4,5,6,7,8,9,10,11,12,13,14,15);
}

// B-fragment (16-bit, 32x16): lane l = col (l&15); k-runs {b..b+7} u {b+16..b+23}
__device__ __forceinline__ v16bf load_b_frag(const __bf16* colbase, int lane, int k0) {
  const __bf16* p = colbase + k0 + ((lane & 16) ? 8 : 0);
  v8bf lo = *(const v8bf*)(p);
  v8bf hi = *(const v8bf*)(p + 16);
  return __builtin_shufflevector(lo, hi, 0,1,2,3,4,5,6,7,8,9,10,11,12,13,14,15);
}

// ---------------------------------------------------------------------------
// TDM: DMA a 512x128 bf16 weight blob from global into LDS offset 0, inserting
// 8 bf16 (4 dwords) of padding after every 128 elems (64 dwords) -> row stride
// 136.  D# per CDNA5 ISA.
// ---------------------------------------------------------------------------
#if __has_builtin(__builtin_amdgcn_tensor_load_to_lds)
#define USE_TDM 1
__device__ __forceinline__ void tdm_load_weights_padded(const void* gsrc) {
  const unsigned long long ga = (unsigned long long)(uintptr_t)gsrc;
  u32x4 g0;
  g0[0] = 1u;                                      // count=1, no gather
  g0[1] = 0u;                                      // lds_addr = base of dyn LDS
  g0[2] = (unsigned)(ga & 0xffffffffull);          // global_addr[31:0]
  g0[3] = (unsigned)((ga >> 32) & 0x01ffffffull)   // global_addr[56:32]
        | (2u << 30);                              // type=2 (image)
  i32x8 g1;
  g1[0] = (int)((1u << 16)     // data_size = 2B
              | (1u << 20)     // pad_enable
              | (5u << 22)     // pad_interval: 64 dwords (= one 128-elem row)
              | (3u << 25));   // pad_amount: 4 dwords (= 8 bf16)
  g1[1] = (int)(128u << 16);   // tensor_dim0 = 128 elems/row (low 16)
  g1[2] = (int)(512u << 16);   // tensor_dim0 hi=0 | tensor_dim1 = 512 rows
  g1[3] = (int)(128u << 16);   // tensor_dim1 hi=0 | tile_dim0 = 128
  g1[4] = 512;                 // tile_dim1 = 512 | tile_dim2 = 0
  g1[5] = 128;                 // tensor_dim0_stride[31:0] = 128
  g1[6] = 0;                   // stride hi | tensor_dim1_stride lo
  g1[7] = 0;
  i32x4 z4 = {0, 0, 0, 0};
#if defined(__clang_major__) && __clang_major__ >= 23
  i32x8 z8 = {0, 0, 0, 0, 0, 0, 0, 0};
  __builtin_amdgcn_tensor_load_to_lds(g0, g1, z4, z4, z8, 0);
#else
  __builtin_amdgcn_tensor_load_to_lds(g0, g1, z4, z4, 0);
#endif
#if __has_builtin(__builtin_amdgcn_s_wait_tensorcnt)
  __builtin_amdgcn_s_wait_tensorcnt(0);
#else
  asm volatile("s_wait_tensorcnt 0x0" ::: "memory");
#endif
}
#else
#define USE_TDM 0
#endif

// fetch weight blob (bf16, row-major 512x128) into padded LDS [512][136]
__device__ __forceinline__ void stage_weights(__bf16* lds, const __bf16* gsrc,
                                              int tid, int nthreads) {
#if USE_TDM
  if (tid < 32) tdm_load_weights_padded(gsrc);   // one wave drives the TDM
  (void)nthreads;
#else
  for (int i = tid; i < kG * kE / 8; i += nthreads) {
    const int row = (i * 8) / kE, k = (i * 8) % kE;
    *(v8bf*)(lds + row * kP + k) = *(const v8bf*)(gsrc + i * 8);
  }
#endif
  __syncthreads();
}

// ---------------------------------------------------------------------------
// Kernel 0: f32 -> bf16 weight conversion (w_ih then w_hh, 65536 elems each)
// ---------------------------------------------------------------------------
__global__ void k_cvt(const float* __restrict__ w_ih, const float* __restrict__ w_hh,
                      __bf16* __restrict__ wih_bf, __bf16* __restrict__ whh_bf) {
  const int i = blockIdx.x * blockDim.x + threadIdx.x;   // 65536 threads
  if (i < kG * kE) {
    wih_bf[i] = (__bf16)w_ih[i];
    whh_bf[i] = (__bf16)w_hh[i];
  }
}

// ---------------------------------------------------------------------------
// Kernel 1: embedding gather + input projection GEMM (32768 x 512 x 128)
//   grid 512 x 128 threads (4 waves); one 16-row tile per wave.
// ---------------------------------------------------------------------------
__global__ void k_embed_project(const int* __restrict__ tokens,
                                const float* __restrict__ emb,
                                const __bf16* __restrict__ wih_bf,
                                const float* __restrict__ b_ih,
                                const float* __restrict__ b_hh,
                                float* __restrict__ P) {
  extern __shared__ char smem[];
  __bf16* wih = (__bf16*)smem;                        // [512][kP]
  const int tid  = threadIdx.x;
  const int wave = tid >> 5;
  const int lane = tid & 31;
  __bf16* estage = wih + kWLDS + wave * (16 * kP);    // [16][kP] per wave

  stage_weights(wih, wih_bf, tid, blockDim.x);

  const int tile    = blockIdx.x * 4 + wave;          // 2048 tiles, exact
  const int rowBase = tile * 16;

  // gather 16 embedding rows into per-wave LDS (f32 -> bf16)
  for (int i = lane; i < 16 * kE / 4; i += 32) {      // 512 float4 groups
    const int row = i >> 5;
    const int k   = (i & 31) * 4;
    const int tok = tokens[rowBase + row];
    float4 v = *(const float4*)(emb + (size_t)tok * kE + k);
    __bf16* d = estage + row * kP + k;
    d[0] = (__bf16)v.x; d[1] = (__bf16)v.y; d[2] = (__bf16)v.z; d[3] = (__bf16)v.w;
  }
  // per-wave staging: in-order DS within the wave, no barrier needed

  const int col  = lane & 15;
  const int mOff = (lane & 16) ? 8 : 0;
  const __bf16* arow = estage + (lane & 15) * kP;

  for (int n0 = 0; n0 < kG; n0 += 16) {
    const float bv = b_ih[n0 + col] + b_hh[n0 + col];  // overlaps the k-loop
    v8f acc;
    for (int r = 0; r < 8; ++r) acc[r] = 0.0f;         // inline SRC2 = 0
    for (int k0 = 0; k0 < kE; k0 += 32) {
      v16bf a = load_a_frag(arow, lane, k0);
      v16bf b = load_b_frag(wih + (n0 + col) * kP, lane, k0);
      acc = __builtin_amdgcn_wmma_f32_16x16x32_bf16(false, a, false, b,
                                                    (short)0, acc, false, false);
    }
    for (int r = 0; r < 8; ++r)
      P[(size_t)(rowBase + r + mOff) * kG + n0 + col] = acc[r] + bv;
  }
}

// ---------------------------------------------------------------------------
// Kernel 2: fused 6-step LSTM over 32448 rows.
//   grid 507 x 128 threads (4 waves); one 16-row tile per wave (2028 exact).
// ---------------------------------------------------------------------------
__global__ void k_lstm(const float* __restrict__ P,
                       const __bf16* __restrict__ whh_bf,
                       float* __restrict__ hlast) {
  extern __shared__ char smem[];
  __bf16* whh = (__bf16*)smem;                        // [512][kP]
  const int tid  = threadIdx.x;
  const int wave = tid >> 5;
  const int lane = tid & 31;
  __bf16* hbuf = whh + kWLDS + wave * (2 * 16 * kP);  // double-buffered h tile

  stage_weights(whh, whh_bf, tid, blockDim.x);

  const int tile    = blockIdx.x * 4 + wave;
  const int rowBase = tile * 16;
  const int col  = lane & 15;
  const int mOff = (lane & 16) ? 8 : 0;

  size_t prow[8];
  for (int r = 0; r < 8; ++r) {
    const int n = rowBase + r + mOff;
    const int b = n / kW;
    prow[r] = (size_t)b * kS + (n - b * kW);
  }

  {  // zero both h buffers (16*kP elems each); c state lives in registers
    v8bf z; for (int e = 0; e < 8; ++e) z[e] = (__bf16)0.0f;
    for (int i = lane; i < 2 * 16 * kP / 8; i += 32) ((v8bf*)hbuf)[i] = z;
  }
  v8f cst[8];
  for (int j = 0; j < 8; ++j)
    for (int r = 0; r < 8; ++r) cst[j][r] = 0.0f;

  for (int t = 0; t < kFW; ++t) {
    __bf16* cur = hbuf + ((t & 1) ? 16 * kP : 0);
    __bf16* nxt = hbuf + ((t & 1) ? 0 : 16 * kP);

    v16bf afr[4];   // h A-fragments, shared by all 32 gate tiles this step
    for (int kk = 0; kk < 4; ++kk)
      afr[kk] = load_a_frag(cur + (lane & 15) * kP, lane, kk * 32);

    if (t + 1 < kFW)  // global_prefetch_b8 of next step's P rows
      __builtin_prefetch(P + (prow[0] + t + 1) * kG, 0, 1);

    for (int j = 0; j < 8; ++j) {
      // P-row gate terms: issued up front, consumed only in the epilogue,
      // so they overlap the 16 WMMAs below.
      v8f pvi, pvf, pvg, pvo;
      for (int r = 0; r < 8; ++r) {
        const float* prp = P + (prow[r] + t) * kG + j * 16 + col;
        pvi[r] = prp[0];
        pvf[r] = prp[kH];
        pvg[r] = prp[2 * kH];
        pvo[r] = prp[3 * kH];
      }
      v8f gi, gf, gg, go;
      for (int r = 0; r < 8; ++r) { gi[r] = 0.f; gf[r] = 0.f; gg[r] = 0.f; go[r] = 0.f; }
      for (int kk = 0; kk < 4; ++kk) {
        const int k0 = kk * 32;
        v16bf bi = load_b_frag(whh + (0 * kH + j * 16 + col) * kP, lane, k0);
        gi = __builtin_amdgcn_wmma_f32_16x16x32_bf16(false, afr[kk], false, bi,
                                                     (short)0, gi, false, false);
        v16bf bf = load_b_frag(whh + (1 * kH + j * 16 + col) * kP, lane, k0);
        gf = __builtin_amdgcn_wmma_f32_16x16x32_bf16(false, afr[kk], false, bf,
                                                     (short)0, gf, false, false);
        v16bf bg = load_b_frag(whh + (2 * kH + j * 16 + col) * kP, lane, k0);
        gg = __builtin_amdgcn_wmma_f32_16x16x32_bf16(false, afr[kk], false, bg,
                                                     (short)0, gg, false, false);
        v16bf bo = load_b_frag(whh + (3 * kH + j * 16 + col) * kP, lane, k0);
        go = __builtin_amdgcn_wmma_f32_16x16x32_bf16(false, afr[kk], false, bo,
                                                     (short)0, go, false, false);
      }
      for (int r = 0; r < 8; ++r) {
        const float cv = sigm(gf[r] + pvf[r]) * cst[j][r]
                       + sigm(gi[r] + pvi[r]) * tanhf(gg[r] + pvg[r]);
        cst[j][r] = cv;
        const float hv = sigm(go[r] + pvo[r]) * tanhf(cv);
        nxt[(r + mOff) * kP + j * 16 + col] = (__bf16)hv;
        if (t == kFW - 1)
          hlast[(size_t)(rowBase + r + mOff) * kH + j * 16 + col] = hv;
      }
    }
  }
}

// ---------------------------------------------------------------------------
// Kernel 3: max over 507 windows per (b,h), then feat @ fc_w^T + fc_b.
// ---------------------------------------------------------------------------
__global__ void k_pool_fc(const float* __restrict__ hlast,
                          const float* __restrict__ fc_w,
                          const float* __restrict__ fc_b,
                          float* __restrict__ out) {
  __shared__ float feat[kH];
  __shared__ float red[kH];
  const int b = blockIdx.x;
  const int h = threadIdx.x;
  float m = -3.4e38f;
  const float* src = hlast + (size_t)b * kW * kH + h;
  for (int w = 0; w < kW; ++w) m = fmaxf(m, src[(size_t)w * kH]);
  feat[h] = m;
  __syncthreads();
  for (int c = 0; c < 2; ++c) {
    red[h] = feat[h] * fc_w[c * kH + h];
    __syncthreads();
    for (int s = kH / 2; s > 0; s >>= 1) {
      if (h < s) red[h] += red[h + s];
      __syncthreads();
    }
    if (h == 0) out[b * 2 + c] = red[0] + fc_b[c];
    __syncthreads();
  }
}

// ---------------------------------------------------------------------------
extern "C" void kernel_launch(void* const* d_in, const int* in_sizes, int n_in,
                              void* d_out, int out_size, void* d_ws, size_t ws_size,
                              hipStream_t stream) {
  const int*   tokens = (const int*)  d_in[0];   // (64,512)
  // d_in[1] = lengths : unused by the reference computation
  const float* emb    = (const float*)d_in[2];   // (50000,128)
  const float* w_ih   = (const float*)d_in[3];   // (512,128)
  const float* w_hh   = (const float*)d_in[4];   // (512,128)
  const float* b_ih   = (const float*)d_in[5];   // (512,)
  const float* b_hh   = (const float*)d_in[6];   // (512,)
  const float* fc_w   = (const float*)d_in[7];   // (2,128)
  const float* fc_b   = (const float*)d_in[8];   // (2,)
  float* out = (float*)d_out;                    // (64,2)

  // workspace: P (64MB) | hlast (16.6MB) | wih_bf | whh_bf
  char* ws = (char*)d_ws;
  float*  P      = (float*)ws;
  float*  hlast  = (float*)(ws + (size_t)kB * kS * kG * sizeof(float));
  __bf16* wih_bf = (__bf16*)(ws + (size_t)kB * kS * kG * sizeof(float)
                                + (size_t)kN * kH * sizeof(float));
  __bf16* whh_bf = wih_bf + (size_t)kG * kE;

  const size_t shm1 = (size_t)(kWLDS + 4 * 16 * kP) * sizeof(__bf16);      // 156672
  const size_t shm2 = (size_t)(kWLDS + 4 * 2 * 16 * kP) * sizeof(__bf16);  // 174080
  (void)hipFuncSetAttribute((const void*)k_embed_project,
                            hipFuncAttributeMaxDynamicSharedMemorySize, (int)shm1);
  (void)hipFuncSetAttribute((const void*)k_lstm,
                            hipFuncAttributeMaxDynamicSharedMemorySize, (int)shm2);

  k_cvt<<<dim3(kG * kE / 256), dim3(256), 0, stream>>>(w_ih, w_hh, wih_bf, whh_bf);
  k_embed_project<<<dim3(kB * kS / 64), dim3(128), shm1, stream>>>(
      tokens, emb, wih_bf, b_ih, b_hh, P);
  k_lstm<<<dim3(kN / 64), dim3(128), shm2, stream>>>(P, whh_bf, hlast);
  k_pool_fc<<<dim3(kB), dim3(kH), 0, stream>>>(hlast, fc_w, fc_b, out);
  (void)in_sizes; (void)n_in; (void)out_size; (void)ws_size;
}
